// stdpNetwork_58050777973142
// MI455X (gfx1250) — matrix operand order repaired and checked by reading, not compiled, not run on hardware
//
#include <hip/hip_runtime.h>

// ---------------- dimensions ----------------
#define T15 15
#define C1 6
#define H0 256
#define F1 30
#define HP 131            // pooled(129) + pad 1
#define F2 250
#define H2 129
#define HW2 (H2*H2)       // 16641
#define K2TOT 270         // 30*3*3
#define K2PAD 288         // 9 chunks of 32
#define F2PAD 256         // 16 tiles of 16
#define BSTRIDE 296       // LDS row stride in halves: 148 dwords -> conflict-free
#define SPKCNT (T15*F2*HW2)   // 62,403,750
#define FHW (F2*HW2)          // 4,160,250

typedef _Float16 half8  __attribute__((ext_vector_type(8)));
typedef _Float16 half16 __attribute__((ext_vector_type(16)));
typedef float    float8 __attribute__((ext_vector_type(8)));

// ---------------- conv1 + fire(15) -> binary spikes ----------------
__global__ __launch_bounds__(256)
void conv1_fire(const float* __restrict__ in, const float* __restrict__ w1,
                float* __restrict__ spk1) {
  const int t = blockIdx.z / F1;
  const int o = blockIdx.z % F1;
  const int h = blockIdx.y * 16 + (threadIdx.x >> 4);
  const int w = blockIdx.x * 16 + (threadIdx.x & 15);
  const float* wt = w1 + o * (C1 * 25);
  const float* it = in + t * (C1 * H0 * H0);
  float acc = 0.f;
  for (int c = 0; c < C1; ++c) {
    #pragma unroll
    for (int kh = 0; kh < 5; ++kh) {
      const int r = h + kh - 2;
      if (r < 0 || r >= H0) continue;
      #pragma unroll
      for (int kw = 0; kw < 5; ++kw) {
        const int cc = w + kw - 2;
        if (cc < 0 || cc >= H0) continue;
        acc += it[c * (H0 * H0) + r * H0 + cc] * wt[c * 25 + kh * 5 + kw];
      }
    }
  }
  spk1[((t * F1 + o) * H0 + h) * H0 + w] = (acc > 15.0f) ? 1.0f : 0.0f;
}

// ---------------- maxpool 2x2 s2 pad1, then pad-by-1 -> [15,30,131,131] ----------------
__global__ __launch_bounds__(256)
void pool_pad(const float* __restrict__ spk1, float* __restrict__ spool) {
  const int idx = blockIdx.x * blockDim.x + threadIdx.x;
  const int total = T15 * F1 * HP * HP;
  if (idx >= total) return;
  const int pw = idx % HP;
  int tmp = idx / HP;
  const int ph = tmp % HP; tmp /= HP;
  const int f = tmp % F1;
  const int t = tmp / F1;
  float v = 0.f;
  if (ph >= 1 && ph <= 129 && pw >= 1 && pw <= 129) {
    const int p = ph - 1, q = pw - 1;
    const float* base = spk1 + (t * F1 + f) * (H0 * H0);
    #pragma unroll
    for (int dr = 0; dr < 2; ++dr) {
      const int r = 2 * p - 1 + dr;
      if (r < 0 || r >= H0) continue;
      #pragma unroll
      for (int dc = 0; dc < 2; ++dc) {
        const int c = 2 * q - 1 + dc;
        if (c < 0 || c >= H0) continue;
        v = fmaxf(v, base[r * H0 + c]);
      }
    }
  }
  spool[idx] = v;
}

// ---------------- pointwise inhibition on binary pooled spikes (in place) ----------------
__global__ __launch_bounds__(256)
void inhibit_pool(float* __restrict__ sp) {
  const int idx = blockIdx.x * blockDim.x + threadIdx.x;
  const int HW = HP * HP;
  if (idx >= HW) return;
  int maxi[T15]; int fired[T15]; int nf = 0;
  for (int t = 0; t < T15; ++t) {
    float mv = -1.f; int mi = 0;
    for (int f = 0; f < F1; ++f) {
      const float x = sp[(t * F1 + f) * HW + idx];
      if (x > mv) { mv = x; mi = f; }
    }
    maxi[t] = mi;
    fired[t] = (mv > 0.f) ? 1 : 0;
    nf += fired[t];
  }
  int e = T15 - nf; e = e < 0 ? 0 : (e > T15 - 1 ? T15 - 1 : e);
  const int win = maxi[e];
  const int valid = fired[T15 - 1];
  for (int t = 0; t < T15; ++t)
    for (int f = 0; f < F1; ++f) {
      const int a = (t * F1 + f) * HW + idx;
      const float x = sp[a];
      sp[a] = (valid && f == win) ? x : 0.f;
    }
}

// ---------------- pack w2 f32[250,270] -> f16[256,288] zero-padded ----------------
__global__ __launch_bounds__(256)
void pack_w2(const float* __restrict__ w2, _Float16* __restrict__ wpack) {
  const int idx = blockIdx.x * blockDim.x + threadIdx.x;
  if (idx >= F2PAD * K2PAD) return;
  const int o = idx / K2PAD;
  const int k = idx - o * K2PAD;
  const float v = (o < F2 && k < K2TOT) ? w2[o * K2TOT + k] : 0.f;
  wpack[idx] = (_Float16)v;
}

// ---------------- conv2: implicit GEMM on WMMA f16->f32, fused fire(2) ----------------
// Block = 256 threads = 8 waves; one (t, h, 16-col tile) per block; wave i owns
// feature tile mt = blockIdx.y*8 + i. B im2col tile staged once in LDS (k-contiguous),
// A fragments are two unconditional global b128 loads from the packed-f16 weights.
__global__ __launch_bounds__(256)
void conv2_wmma(const float* __restrict__ spk_in, const _Float16* __restrict__ wpack,
                float* __restrict__ pot_out) {
  __shared__ __attribute__((aligned(16))) _Float16 ldsB[16 * BSTRIDE];

  const int wtile = blockIdx.x % 9;
  const int h     = blockIdx.x / 9;      // 0..128
  const int t     = blockIdx.z;
  const int wbase = wtile * 16;

  const float* inT = spk_in + t * (F1 * HP * HP);

  // ---- fill LDS B tile: ldsB[n*BSTRIDE + k] = im2col(t,h,wbase+n,k), zero-padded ----
  for (int e = threadIdx.x; e < 16 * K2PAD; e += 256) {
    const int n = e & 15;          // lanes vary n -> coalesced global reads
    const int k = e >> 4;
    const int wcol = wbase + n;
    float v = 0.f;
    if (k < K2TOT && wcol < H2) {
      const int ci = k / 9, rem = k % 9;
      const int kh = rem / 3, kw = rem % 3;
      v = inT[ci * (HP * HP) + (h + kh) * HP + (wcol + kw)];
    }
    ldsB[n * BSTRIDE + k] = (_Float16)v;
  }
  __syncthreads();

  const int lane = threadIdx.x & 31;
  const int wave = threadIdx.x >> 5;
  const int n16  = lane & 15;
  const int grp  = lane >> 4;
  const int mt   = blockIdx.y * 8 + wave;

  const _Float16* Abase = wpack + (mt * 16 + n16) * K2PAD + 8 * grp;
  const _Float16* Bbase = &ldsB[n16 * BSTRIDE + 8 * grp];

  float8 acc = {};
  #pragma unroll
  for (int kc = 0; kc < 9; ++kc) {
    const int kbase = kc * 32;
    // A fragment: halves k = kbase+8g+[0..8) and kbase+8g+16+[0..8)
    const half8* ap = (const half8*)(Abase + kbase);
    const half8 a0 = ap[0];
    const half8 a1 = ap[2];
    // B fragment: halves k = kbase+8g+[0..8) and kbase+8g+16+[0..8)
    const half8* bp = (const half8*)(Bbase + kbase);
    const half8 b0 = bp[0];
    const half8 b1 = bp[2];
    const half16 a = __builtin_shufflevector(a0, a1, 0, 1, 2, 3, 4, 5, 6, 7,
                                             8, 9, 10, 11, 12, 13, 14, 15);
    const half16 b = __builtin_shufflevector(b0, b1, 0, 1, 2, 3, 4, 5, 6, 7,
                                             8, 9, 10, 11, 12, 13, 14, 15);
    acc = __builtin_amdgcn_wmma_f32_16x16x32_f16(false, a, false, b,
                                                 (short)0, acc, false, false);
  }

  // ---- epilogue: fire(thr=2), store pot ----
  const int wcol = wbase + n16;
  #pragma unroll
  for (int v = 0; v < 8; ++v) {
    const int oo = mt * 16 + v + grp * 8;
    if (oo < F2 && wcol < H2) {
      const float p  = acc[v];
      const float pf = (p > 2.0f) ? p : 0.0f;
      pot_out[((t * F2 + oo) * H2 + h) * H2 + wcol] = pf;
    }
  }
}

// ---------------- pointwise inhibition on pot2: phase 1 (winner per location) ----------------
__global__ __launch_bounds__(256)
void inhibit_pot_p1(const float* __restrict__ pot, int* __restrict__ winmap) {
  const int idx = blockIdx.x * blockDim.x + threadIdx.x;
  if (idx >= HW2) return;
  int maxi[T15]; int fired[T15]; int nf = 0;
  for (int t = 0; t < T15; ++t) {
    float mv = -1.f; int mi = 0;
    for (int f = 0; f < F2; ++f) {
      const float x = pot[(t * F2 + f) * HW2 + idx];
      if (x > mv) { mv = x; mi = f; }
    }
    maxi[t] = mi;
    fired[t] = (mv > 0.f) ? 1 : 0;
    nf += fired[t];
  }
  int e = T15 - nf; e = e < 0 ? 0 : (e > T15 - 1 ? T15 - 1 : e);
  winmap[idx] = fired[T15 - 1] ? maxi[e] : -1;
}

// ---------------- phase 2: apply coef, write final pot (in place) and spk ----------------
__global__ __launch_bounds__(256)
void inhibit_pot_p2(float* __restrict__ pot, float* __restrict__ spk,
                    const int* __restrict__ winmap) {
  const int idx = blockIdx.x * blockDim.x + threadIdx.x;
  if (idx >= SPKCNT) return;
  const int loc = idx % HW2;
  const int f   = (idx / HW2) % F2;
  const int win = winmap[loc];
  const float p  = pot[idx];
  const float po = (f == win) ? p : 0.f;
  pot[idx] = po;
  spk[idx] = (po > 0.f) ? 1.f : 0.f;
}

// ---------------- k-winners ----------------
__global__ __launch_bounds__(1)
void kw_init(unsigned int* __restrict__ maxtr, unsigned long long* __restrict__ slot) {
  *maxtr = 0u;
  *slot  = 0ull;
}

__global__ __launch_bounds__(256)
void kw_prepare(const float* __restrict__ spk, const float* __restrict__ pot,
                float* __restrict__ wval, float* __restrict__ wnspk,
                unsigned int* __restrict__ maxtr) {
  const int idx = blockIdx.x * blockDim.x + threadIdx.x;
  if (idx >= FHW) return;
  int ns = 0;
  for (int t = 0; t < T15; ++t) ns += (spk[t * FHW + idx] > 0.f) ? 1 : 0;
  int e = T15 - ns; e = e < 0 ? 0 : (e > T15 - 1 ? T15 - 1 : e);
  const float val = pot[e * FHW + idx];
  wval[idx]  = val;
  wnspk[idx] = (float)ns;
  const float tr = (ns > 0) ? val : 0.f;     // nonneg: uint bit compare == float compare
  atomicMax(maxtr, __float_as_uint(tr));
}

__global__ __launch_bounds__(256)
void kw_total(float* __restrict__ wval, const float* __restrict__ wnspk,
              const unsigned int* __restrict__ maxtr) {
  const int idx = blockIdx.x * blockDim.x + threadIdx.x;
  if (idx >= FHW) return;
  const float v  = __uint_as_float(*maxtr) * (float)T15;
  const float ns = wnspk[idx];
  wval[idx] = ns * (wval[idx] + v);          // total = nspk*(values + v)
}

__global__ __launch_bounds__(1)
void kw_reset(unsigned long long* __restrict__ slot) { *slot = 0ull; }

__global__ __launch_bounds__(256)
void kw_argmax(const float* __restrict__ total, unsigned long long* __restrict__ slot) {
  __shared__ unsigned long long red[256];
  unsigned long long best = 0ull;
  for (int i = blockIdx.x * blockDim.x + threadIdx.x; i < FHW;
       i += gridDim.x * blockDim.x) {
    const float v = total[i];
    const unsigned long long p =
        ((unsigned long long)__float_as_uint(v) << 32) | (unsigned int)i;
    if (p > best) best = p;
  }
  red[threadIdx.x] = best;
  __syncthreads();
  for (int s = 128; s > 0; s >>= 1) {
    if ((int)threadIdx.x < s) {
      const unsigned long long o = red[threadIdx.x + s];
      if (o > red[threadIdx.x]) red[threadIdx.x] = o;
    }
    __syncthreads();
  }
  if (threadIdx.x == 0) atomicMax(slot, red[0]);
}

__global__ __launch_bounds__(256)
void kw_suppress(float* __restrict__ total, const unsigned long long* __restrict__ slot,
                 float* __restrict__ winners, int iter) {
  const unsigned long long s = *slot;
  const float bv = __uint_as_float((unsigned int)(s >> 32));
  const int bidx = (int)(s & 0xffffffffu);
  const int wf = bidx / HW2;
  const int rr = bidx % HW2;
  const int wr = rr / H2, wc = rr % H2;
  const bool valid = (bv != 0.f);
  const int gid = blockIdx.x * blockDim.x + threadIdx.x;
  if (gid == 0) {
    winners[iter * 3 + 0] = valid ? (float)wf : -1.f;
    winners[iter * 3 + 1] = valid ? (float)wr : -1.f;
    winners[iter * 3 + 2] = valid ? (float)wc : -1.f;
  }
  if (!valid || gid >= FHW) return;
  const int f = gid / HW2;
  const int l = gid % HW2;
  const int r = l / H2, c = l % H2;
  const bool sup = (f == wf) || ((abs(r - wr) <= 1) && (abs(c - wc) <= 1));
  if (sup) total[gid] = 0.f;
}

// ---------------- host-side launcher ----------------
extern "C" void kernel_launch(void* const* d_in, const int* in_sizes, int n_in,
                              void* d_out, int out_size, void* d_ws, size_t ws_size,
                              hipStream_t stream) {
  const float* in = (const float*)d_in[0];   // [15,6,256,256]
  const float* w1 = (const float*)d_in[1];   // [30,6,5,5]
  const float* w2 = (const float*)d_in[2];   // [250,30,3,3] == [250,270]
  // d_in[3] = max_layer (==2 per setup); hardcoded path.

  float* out    = (float*)d_out;
  float* spkOut = out;                       // [15,250,129,129]
  float* potOut = out + SPKCNT;              // [15,250,129,129]
  float* winOut = out + 2 * (size_t)SPKCNT;  // [8,3]

  char* ws = (char*)d_ws;
  float*    spk1  = (float*)   (ws);              // 117,964,800 B
  float*    spool = (float*)   (ws + 117964800);  //  30,889,800 B
  int*      winmp = (int*)     (ws + 148854608);  //      66,564 B
  float*    wval  = (float*)   (ws + 148921216);  //  16,641,000 B
  float*    wnspk = (float*)   (ws + 165562240);  //  16,641,000 B
  _Float16* wpack = (_Float16*)(ws + 182203264);  //     147,456 B
  unsigned int*       maxtr = (unsigned int*)      (ws + 182350720);
  unsigned long long* slot  = (unsigned long long*)(ws + 182350728);

  // 0) pack weights for WMMA
  pack_w2<<<(F2PAD * K2PAD + 255) / 256, 256, 0, stream>>>(w2, wpack);

  // 1) conv1 + fire(15)
  conv1_fire<<<dim3(16, 16, T15 * F1), 256, 0, stream>>>(in, w1, spk1);

  // 2) pool + pad
  {
    const int n = T15 * F1 * HP * HP;
    pool_pad<<<(n + 255) / 256, 256, 0, stream>>>(spk1, spool);
  }

  // 3) pointwise inhibition on pooled spikes
  inhibit_pool<<<(HP * HP + 255) / 256, 256, 0, stream>>>(spool);

  // 4) conv2 via WMMA + fire(2) -> potOut (raw fired pot)
  conv2_wmma<<<dim3(9 * H2, 2, T15), 256, 0, stream>>>(spool, wpack, potOut);

  // 5) pointwise inhibition on pot, spk = sign(pot)
  inhibit_pot_p1<<<(HW2 + 255) / 256, 256, 0, stream>>>(potOut, winmp);
  inhibit_pot_p2<<<(SPKCNT + 255) / 256, 256, 0, stream>>>(potOut, spkOut, winmp);

  // 6) k-winners
  kw_init<<<1, 1, 0, stream>>>(maxtr, slot);
  kw_prepare<<<(FHW + 255) / 256, 256, 0, stream>>>(spkOut, potOut, wval, wnspk, maxtr);
  kw_total<<<(FHW + 255) / 256, 256, 0, stream>>>(wval, wnspk, maxtr);
  for (int iter = 0; iter < 8; ++iter) {
    kw_reset<<<1, 1, 0, stream>>>(slot);
    kw_argmax<<<2048, 256, 0, stream>>>(wval, slot);
    kw_suppress<<<(FHW + 255) / 256, 256, 0, stream>>>(wval, slot, winOut, iter);
  }
}
